// DL_16638703305357
// MI455X (gfx1250) — matrix operand order repaired and checked by reading, not compiled, not run on hardware
//
#include <hip/hip_runtime.h>
#include <hip/hip_bf16.h>

typedef __attribute__((ext_vector_type(16))) _Float16 v16h;
typedef __attribute__((ext_vector_type(8)))  float    v8f;

#define B_    8
#define CIN_  128
#define COUT_ 128
#define HID_  64
#define H_    128
#define W_    128
#define HW_   (H_ * W_)
#define XS_STRIDE 144   // halfs; multiple of 16 -> 32B-aligned rows for v16h LDS loads

__device__ __forceinline__ float silu_f(float v) {
    return v / (1.0f + __expf(-v));
}

// ---- CDNA5 async global->LDS copy (ASYNCcnt path), 16B granules -----------
__device__ __forceinline__ void async_copy_b128(void* lds_dst, const void* gsrc) {
    unsigned lds_off = (unsigned)(size_t)lds_dst;             // LDS addr = addr[31:0]
    unsigned long long ga = (unsigned long long)(size_t)gsrc; // 64-bit global address
    asm volatile("global_load_async_to_lds_b128 %0, %1, off"
                 :: "v"(lds_off), "v"(ga) : "memory");
}
__device__ __forceinline__ void async_wait0() {
    asm volatile("s_wait_asynccnt 0x0" ::: "memory");
}

// ---------------------------------------------------------------------------
// Kernel 1: fused dual 1x1 conv (x -> x1, x2) with WMMA f16 (fp32 accum)
// Block = 256 threads = 8 waves. Block tile: 64 pixels, all 128 output rows
// (waves 0-3 -> w1 rows 0..63, waves 4-7 -> w2 rows 0..63).
// ---------------------------------------------------------------------------
__global__ __launch_bounds__(256) void k_dual_conv1x1(
    const float* __restrict__ x,
    const float* __restrict__ w1, const float* __restrict__ s1, const float* __restrict__ b1,
    const float* __restrict__ w2, const float* __restrict__ s2, const float* __restrict__ b2,
    float* __restrict__ x1o, float* __restrict__ x2o)
{
    __shared__ _Float16 xs[64 * XS_STRIDE];   // [pixel][channel] f16

    const int b     = blockIdx.y;
    const int pbase = blockIdx.x * 64;
    const int tid   = threadIdx.x;
    const int lane  = tid & 31;
    const int wv    = tid >> 5;               // 0..7

    // ---- stage x tile to LDS as f16, [p][c] ----
    {
        const int p  = tid & 63;
        const int c0 = tid >> 6;              // 0..3
        const float* xb = x + ((size_t)b * CIN_) * HW_ + pbase;
        #pragma unroll 4
        for (int c = c0; c < CIN_; c += 4) {
            float v = xb[(size_t)c * HW_ + p];
            __builtin_prefetch(&xb[(size_t)c * HW_ + p + 64], 0, 1); // next tile
            xs[p * XS_STRIDE + c] = (_Float16)v;
        }
    }
    __syncthreads();

    // ---- per-wave weight / output selection ----
    const bool second = (wv >= 4);
    const int  mt     = second ? (wv - 4) : wv;       // M-tile 0..3
    const float* wm   = second ? w2 : w1;
    const float* sc   = second ? s2 : s1;
    const float* bi   = second ? b2 : b1;
    float*       outp = second ? x2o : x1o;

    // ---- A fragments: 16x32 f16 layout (row = lane&15) ----
    const int row   = mt * 16 + (lane & 15);
    const int abase = (lane < 16) ? 0 : 16;
    v16h a[4];
    #pragma unroll
    for (int kb = 0; kb < 4; ++kb) {
        const float* wrow = wm + (size_t)row * CIN_ + kb * 32 + abase;
        #pragma unroll
        for (int hj = 0; hj < 16; ++hj) {
            const int K = hj;   // 0..15 contiguous (rel.)
            a[kb][hj] = (_Float16)wrow[K];
        }
    }

    const int np    = lane & 15;
    const int kside = (lane < 16) ? 0 : 16;
    const int mofs  = (lane < 16) ? 0 : 8;

    #pragma unroll
    for (int nt = 0; nt < 4; ++nt) {
        const int p = nt * 16 + np;
        v8f acc = {};
        #pragma unroll
        for (int kb = 0; kb < 4; ++kb) {
            v16h bf = *(const v16h*)&xs[p * XS_STRIDE + kb * 32 + kside];
            acc = __builtin_amdgcn_wmma_f32_16x16x32_f16(
                false, a[kb], false, bf, (short)0, acc, false, false);
        }
        #pragma unroll
        for (int r = 0; r < 8; ++r) {
            const int o = mt * 16 + mofs + r;
            float v = acc[r] * sc[o] + bi[o];
            outp[((size_t)b * HID_ + o) * HW_ + pbase + p] = silu_f(v);
        }
    }
}

// ---------------------------------------------------------------------------
// Kernel 2: 3x3 offset conv (x1[64ch] -> 18 offset channels).
// Weights streamed to LDS with CDNA5 async global->LDS copies.
// ---------------------------------------------------------------------------
__global__ __launch_bounds__(256) void k_offset_conv(
    const float* __restrict__ x1,
    const float* __restrict__ off_w, const float* __restrict__ off_b,
    float* __restrict__ off)
{
    __shared__ __align__(16) float wsh[18 * HID_ * 9];   // 10368 floats
    {
        const int nvec = (18 * HID_ * 9) / 4;            // 2592 x b128
        for (int i4 = threadIdx.x; i4 < nvec; i4 += 256)
            async_copy_b128(&wsh[i4 * 4], off_w + i4 * 4);
        async_wait0();
    }
    __syncthreads();

    const int gid = blockIdx.x * 256 + threadIdx.x;    // over B*HW
    const int b   = gid / HW_;
    const int pix = gid - b * HW_;
    const int y   = pix >> 7;
    const int xc  = pix & (W_ - 1);

    float acc[18];
    #pragma unroll
    for (int kk = 0; kk < 18; ++kk) acc[kk] = off_b[kk];

    const float* xb = x1 + (size_t)b * HID_ * HW_;
    for (int ky = 0; ky < 3; ++ky) {
        const int yy = y + ky - 1;
        if (yy < 0 || yy >= H_) continue;
        for (int kx = 0; kx < 3; ++kx) {
            const int xx = xc + kx - 1;
            if (xx < 0 || xx >= W_) continue;
            const int npix = yy * W_ + xx;
            const int t = ky * 3 + kx;
            for (int c = 0; c < HID_; ++c) {
                const float v = xb[(size_t)c * HW_ + npix];
                const float* wp = &wsh[c * 9 + t];
                #pragma unroll
                for (int kk = 0; kk < 18; ++kk)
                    acc[kk] = fmaf(v, wp[kk * HID_ * 9], acc[kk]);
            }
        }
    }
    #pragma unroll
    for (int kk = 0; kk < 18; ++kk)
        off[((size_t)b * 18 + kk) * HW_ + pix] = acc[kk];
}

// ---------------------------------------------------------------------------
// Kernel 3: deformable depthwise sampling. One thread per (b, pixel); corner
// indices/weights computed once per tap, reused across all 64 channels.
// Depthwise weights staged via async global->LDS.
// ---------------------------------------------------------------------------
__global__ __launch_bounds__(256) void k_deform(
    const float* __restrict__ x1, const float* __restrict__ off,
    const float* __restrict__ dw, float* __restrict__ x1d)
{
    __shared__ __align__(16) float dwsh[HID_ * 9];       // 576 floats
    {
        const int nvec = (HID_ * 9) / 4;                 // 144 x b128
        if (threadIdx.x < nvec)
            async_copy_b128(&dwsh[threadIdx.x * 4], dw + threadIdx.x * 4);
        async_wait0();
    }
    __syncthreads();

    const int gid = blockIdx.x * 256 + threadIdx.x;    // over B*HW
    const int b   = gid / HW_;
    const int pix = gid - b * HW_;
    const int y   = pix >> 7;
    const int xc  = pix & (W_ - 1);

    int   idx[9][4];
    float wgt[9][4];
    const float* offb = off + (size_t)b * 18 * HW_;

    #pragma unroll
    for (int t = 0; t < 9; ++t) {
        const float dy = offb[(size_t)(2 * t + 0) * HW_ + pix];
        const float dx = offb[(size_t)(2 * t + 1) * HW_ + pix];
        const float py = (float)(y + t / 3 - 1) + dy;
        const float px = (float)(xc + t % 3 - 1) + dx;
        const float fy = floorf(py), fx = floorf(px);
        const float wy = py - fy,    wx = px - fx;
        const int y0 = (int)fy, x0 = (int)fx;
        const float cw[4] = { (1.f - wy) * (1.f - wx), (1.f - wy) * wx,
                              wy * (1.f - wx),          wy * wx };
        const int cy[4] = { y0, y0, y0 + 1, y0 + 1 };
        const int cx[4] = { x0, x0 + 1, x0, x0 + 1 };
        #pragma unroll
        for (int k = 0; k < 4; ++k) {
            const bool valid = (cy[k] >= 0) && (cy[k] < H_) && (cx[k] >= 0) && (cx[k] < W_);
            const int yi = min(max(cy[k], 0), H_ - 1);
            const int xi = min(max(cx[k], 0), W_ - 1);
            idx[t][k] = yi * W_ + xi;
            wgt[t][k] = valid ? cw[k] : 0.0f;
        }
    }

    const float* xb = x1 + (size_t)b * HID_ * HW_;
    for (int c = 0; c < HID_; ++c) {
        const float* xcptr = xb + (size_t)c * HW_;
        float acc = 0.0f;
        #pragma unroll
        for (int t = 0; t < 9; ++t) {
            float v = wgt[t][0] * xcptr[idx[t][0]]
                    + wgt[t][1] * xcptr[idx[t][1]]
                    + wgt[t][2] * xcptr[idx[t][2]]
                    + wgt[t][3] * xcptr[idx[t][3]];
            acc = fmaf(v, dwsh[c * 9 + t], acc);
        }
        x1d[((size_t)b * HID_ + c) * HW_ + pix] = acc;
    }
}

// ---------------------------------------------------------------------------
// Kernel 4: final 1x1 conv on concat(x1d, x2) -> out, WMMA f16, silu epilogue
// ---------------------------------------------------------------------------
__global__ __launch_bounds__(256) void k_final_conv1x1(
    const float* __restrict__ x1d, const float* __restrict__ x2,
    const float* __restrict__ w3, const float* __restrict__ s3, const float* __restrict__ b3,
    float* __restrict__ out)
{
    __shared__ _Float16 xs[64 * XS_STRIDE];

    const int b     = blockIdx.y;
    const int pbase = blockIdx.x * 64;
    const int tid   = threadIdx.x;
    const int lane  = tid & 31;
    const int wv    = tid >> 5;     // M-tile 0..7 (COUT=128)

    {
        const int p  = tid & 63;
        const int c0 = tid >> 6;
        #pragma unroll 4
        for (int c = c0; c < CIN_; c += 4) {
            const float* src = (c < HID_)
                ? &x1d[((size_t)b * HID_ + c) * HW_ + pbase + p]
                : &x2[((size_t)b * HID_ + (c - HID_)) * HW_ + pbase + p];
            __builtin_prefetch(src + 64, 0, 1);
            xs[p * XS_STRIDE + c] = (_Float16)(*src);
        }
    }
    __syncthreads();

    const int row   = wv * 16 + (lane & 15);
    const int abase = (lane < 16) ? 0 : 16;
    v16h a[4];
    #pragma unroll
    for (int kb = 0; kb < 4; ++kb) {
        const float* wrow = w3 + (size_t)row * CIN_ + kb * 32 + abase;
        #pragma unroll
        for (int hj = 0; hj < 16; ++hj) {
            const int K = hj;
            a[kb][hj] = (_Float16)wrow[K];
        }
    }

    const int np    = lane & 15;
    const int kside = (lane < 16) ? 0 : 16;
    const int mofs  = (lane < 16) ? 0 : 8;

    #pragma unroll
    for (int nt = 0; nt < 4; ++nt) {
        const int p = nt * 16 + np;
        v8f acc = {};
        #pragma unroll
        for (int kb = 0; kb < 4; ++kb) {
            v16h bf = *(const v16h*)&xs[p * XS_STRIDE + kb * 32 + kside];
            acc = __builtin_amdgcn_wmma_f32_16x16x32_f16(
                false, a[kb], false, bf, (short)0, acc, false, false);
        }
        #pragma unroll
        for (int r = 0; r < 8; ++r) {
            const int o = wv * 16 + mofs + r;
            float v = acc[r] * s3[o] + b3[o];
            out[((size_t)b * COUT_ + o) * HW_ + pbase + p] = silu_f(v);
        }
    }
}

// ---------------------------------------------------------------------------
extern "C" void kernel_launch(void* const* d_in, const int* in_sizes, int n_in,
                              void* d_out, int out_size, void* d_ws, size_t ws_size,
                              hipStream_t stream)
{
    const float* x     = (const float*)d_in[0];
    const float* w1    = (const float*)d_in[1];
    const float* s1    = (const float*)d_in[2];
    const float* b1    = (const float*)d_in[3];
    const float* w2    = (const float*)d_in[4];
    const float* s2    = (const float*)d_in[5];
    const float* b2    = (const float*)d_in[6];
    const float* w3    = (const float*)d_in[7];
    const float* s3    = (const float*)d_in[8];
    const float* b3    = (const float*)d_in[9];
    const float* off_w = (const float*)d_in[10];
    const float* off_b = (const float*)d_in[11];
    const float* dw    = (const float*)d_in[12];
    float* out = (float*)d_out;

    float* ws  = (float*)d_ws;
    const size_t nX1 = (size_t)B_ * HID_ * HW_;        // 8,388,608
    float* x1  = ws;
    float* x2  = ws + nX1;
    float* x1d = ws + 2 * nX1;
    float* off = ws + 3 * nX1;                         // 8*18*HW floats

    const dim3 gConv(HW_ / 64, B_);   // 256 x 8
    const dim3 blk(256);

    k_dual_conv1x1<<<gConv, blk, 0, stream>>>(x, w1, s1, b1, w2, s2, b2, x1, x2);
    k_offset_conv<<<(B_ * HW_) / 256, blk, 0, stream>>>(x1, off_w, off_b, off);
    k_deform<<<(B_ * HW_) / 256, blk, 0, stream>>>(x1, off, dw, x1d);
    k_final_conv1x1<<<gConv, blk, 0, stream>>>(x1d, x2, w3, s3, b3, out);
}
